// SelfAttention_11407433138769
// MI455X (gfx1250) — matrix operand order repaired and checked by reading, not compiled, not run on hardware
//
#include <hip/hip_runtime.h>
#include <hip/hip_bf16.h>
#include <math.h>

#define NB 8
#define NN 2048
#define DD 1024
#define TWO_N 4096
#define WPAD 40   // padded LDS row stride (elements) for staged W slabs

typedef __bf16 bf16_t;
typedef __attribute__((ext_vector_type(16))) __bf16 v16bf;
typedef __attribute__((ext_vector_type(8)))  __bf16 v8bf;
typedef __attribute__((ext_vector_type(4)))  __bf16 v4bf;
typedef __attribute__((ext_vector_type(8)))  float  v8f;
typedef __attribute__((ext_vector_type(4)))  float  v4f;
typedef int v4i_vs __attribute__((vector_size(16)));   // matches builtin param

// CDNA5 async global->LDS copy support (guarded; sync fallback otherwise)
#if defined(__has_builtin)
#  if __has_builtin(__builtin_amdgcn_global_load_async_to_lds_b128) && \
      __has_builtin(__builtin_amdgcn_s_wait_asynccnt)
#    define HAVE_ASYNC_LDS 1
#  endif
#endif
#ifndef HAVE_ASYNC_LDS
#  define HAVE_ASYNC_LDS 0
#endif

// ---------------------------------------------------------------------------
// WMMA helper: D = A(16x32 bf16) * B(32x16 bf16) + C(16x16 f32)
// ---------------------------------------------------------------------------
__device__ __forceinline__ v8f wmma_bf16(v16bf a, v16bf b, v8f c) {
  return __builtin_amdgcn_wmma_f32_16x16x32_bf16(
      /*neg_a=*/false, a, /*neg_b=*/false, b,
      /*c_mod=*/(short)0, c, /*reuse_a=*/false, /*reuse_b=*/false);
}

// A-matrix fragment (16x32, 16-bit) from a row-major bf16 matrix.
// Layout: lanes 0-15 hold row M=lane, K = k0..k0+15;
//         lanes 16-31 hold row M=lane-16, K = k0+16..k0+31 (matches B).
__device__ __forceinline__ v16bf load_a_bf16(const bf16_t* __restrict__ base,
                                             int ld, int k0, int lane) {
  int m   = lane & 15;
  int sub = (lane >> 4) * 16;
  const bf16_t* p = base + (size_t)m * ld + k0;
  v8bf lo = *(const v8bf*)(p + sub);
  v8bf hi = *(const v8bf*)(p + sub + 8);
  v16bf r;
#pragma unroll
  for (int i = 0; i < 8; ++i) { r[i] = lo[i]; r[8 + i] = hi[i]; }
  return r;
}

// B-matrix fragment (32x16, 16-bit). Column n of B contiguous in the source:
// lane L holds column (L&15), K = k0 + (L>>4)*16 .. +15 contiguous.
__device__ __forceinline__ v16bf load_b_bf16(const bf16_t* __restrict__ base,
                                             int ld, int k0, int lane) {
  int n    = lane & 15;
  int koff = (lane >> 4) * 16;
  return *(const v16bf*)(base + (size_t)n * ld + k0 + koff);
}

// Per-lane source address of a B fragment (for global_prefetch)
__device__ __forceinline__ const bf16_t* baddr(const bf16_t* base, int ld,
                                               int k0, int lane) {
  return base + (size_t)(lane & 15) * ld + k0 + ((lane >> 4) * 16);
}

// ---------------------------------------------------------------------------
// Kernel 1: circulant bias column  c[t] = real(ifft(R))[t]
// ---------------------------------------------------------------------------
__global__ void dft_kernel(const float* __restrict__ R, float* __restrict__ c) {
  int t = blockIdx.x * blockDim.x + threadIdx.x;
  if (t >= TWO_N) return;
  const float step = 6.283185307179586f / (float)TWO_N;
  float acc = 0.f;
  for (int j = 0; j < TWO_N; ++j) {
    int m = (t * j) & (TWO_N - 1);        // exact angle reduction mod 2pi
    acc += R[j] * cosf(step * (float)m);
  }
  c[t] = acc * (1.0f / (float)TWO_N);
}

// ---------------------------------------------------------------------------
// Kernel 2: Wt[e][d] = (bf16) W[d][e]   (so B-fragments load contiguously)
// ---------------------------------------------------------------------------
__global__ void wtrans_kernel(const float* __restrict__ W,
                              bf16_t* __restrict__ Wt) {
  int idx = blockIdx.x * blockDim.x + threadIdx.x;
  int e = idx >> 10;
  int d = idx & 1023;
  Wt[(size_t)e * DD + d] = (bf16_t)W[(size_t)d * DD + e];
}

// ---------------------------------------------------------------------------
// Kernel 2b: x -> bf16 (one-time, so the GEMM hot loops read bf16 directly)
// ---------------------------------------------------------------------------
__global__ void xconv_kernel(const float* __restrict__ x,
                             bf16_t* __restrict__ xb) {
  size_t i = ((size_t)blockIdx.x * blockDim.x + threadIdx.x) * 4;
  v4f v = *(const v4f*)(x + i);
  v4bf o;
#pragma unroll
  for (int j = 0; j < 4; ++j) o[j] = (bf16_t)v[j];
  *(v4bf*)(xb + i) = o;
}

// ---------------------------------------------------------------------------
// Kernel 3: projection  out = tau * (x @ W)    [16384 x 1024] x [1024 x 1024]
// Block = 128 threads (4 waves); 64x64 output tile per block.
// The W slab (64 cols x 32 k, shared by all 4 waves) is staged into LDS via
// CDNA5 async global->LDS copies, double-buffered one k-step ahead
// (ASYNCcnt-tracked), instead of 4x redundant per-wave global loads.
// vtranspose==1 stores v as [B][D][N] (transposed per batch) for the AV GEMM.
// ---------------------------------------------------------------------------
__global__ void __launch_bounds__(128)
qkv_kernel(const bf16_t* __restrict__ xb, const bf16_t* __restrict__ Wt,
           const float* __restrict__ tau_p, bf16_t* __restrict__ outp,
           int vtranspose) {
  __shared__ bf16_t wbuf[2][64 * WPAD];

  int tid  = threadIdx.x;
  int lane = tid & 31;
  int wave = tid >> 5;
  int row0 = blockIdx.y * 64 + wave * 16;   // global row in [0, B*N)
  int col0 = blockIdx.x * 64;
  const bf16_t* xrow  = xb + (size_t)row0 * DD;
  const bf16_t* wbase = Wt + (size_t)col0 * DD;

  // Producer: thread t copies 32B: row = t>>1, half = (t&1)*16 elements.
  int prow  = tid >> 1;
  int phalf = (tid & 1) * 16;

  auto stage = [&](int slab, int k0) {
    const bf16_t* src = wbase + (size_t)prow * DD + k0 + phalf;
    bf16_t* dst = &wbuf[slab][prow * WPAD + phalf];
#if HAVE_ASYNC_LDS
    __builtin_amdgcn_global_load_async_to_lds_b128(
        (__attribute__((address_space(1))) v4i_vs*)src,
        (__attribute__((address_space(3))) v4i_vs*)dst, 0, 0);
    __builtin_amdgcn_global_load_async_to_lds_b128(
        (__attribute__((address_space(1))) v4i_vs*)(src + 8),
        (__attribute__((address_space(3))) v4i_vs*)(dst + 8), 0, 0);
#else
    *(v8bf*)dst       = *(const v8bf*)src;
    *(v8bf*)(dst + 8) = *(const v8bf*)(src + 8);
#endif
  };
  auto async_wait0 = []() {
#if HAVE_ASYNC_LDS
    __builtin_amdgcn_s_wait_asynccnt(0);
#endif
  };
  // B fragment from the staged LDS slab (two 16B reads, 16B-aligned: WPAD=40)
  auto bfragLDS = [&](int slab, int j) -> v16bf {
    const bf16_t* p = &wbuf[slab][(j * 16 + (lane & 15)) * WPAD +
                                  ((lane >> 4) * 16)];
    v8bf lo = *(const v8bf*)p;
    v8bf hi = *(const v8bf*)(p + 8);
    v16bf r;
#pragma unroll
    for (int i = 0; i < 8; ++i) { r[i] = lo[i]; r[8 + i] = hi[i]; }
    return r;
  };

  stage(0, 0);
  v16bf a = load_a_bf16(xrow, DD, 0, lane);
  async_wait0();
  __syncthreads();
  stage(1, 32);

  v8f acc[4] = {};
  int slab = 0;
  for (int k0 = 0; k0 < DD; k0 += 32) {
    int kn  = (k0 + 32 < DD) ? (k0 + 32) : k0;   // clamped (dup work discarded)
    int kn2 = (k0 + 64 < DD) ? (k0 + 64) : kn;
    v16bf b0 = bfragLDS(slab, 0);
    v16bf b1 = bfragLDS(slab, 1);
    v16bf b2 = bfragLDS(slab, 2);
    v16bf b3 = bfragLDS(slab, 3);
    v16bf an = load_a_bf16(xrow, DD, kn, lane);
    acc[0] = wmma_bf16(a, b0, acc[0]);
    acc[1] = wmma_bf16(a, b1, acc[1]);
    acc[2] = wmma_bf16(a, b2, acc[2]);
    acc[3] = wmma_bf16(a, b3, acc[3]);
    a = an;
    async_wait0();        // copies for the other slab have landed
    __syncthreads();      // everyone done reading `slab`
    stage(slab, kn2);     // refill it one k-step ahead
    slab ^= 1;
  }

  float tau = tau_p[0];
  int n  = lane & 15;
  int mb = (lane >> 4) * 8;
#pragma unroll
  for (int j = 0; j < 4; ++j) {
    int gcol = col0 + j * 16 + n;
#pragma unroll
    for (int r = 0; r < 8; ++r) {
      float vl = acc[j][r] * tau;
      int grow = row0 + mb + r;
      if (!vtranspose) {
        outp[(size_t)grow * DD + gcol] = (bf16_t)vl;
      } else {
        int bb = grow >> 11;              // batch
        int nn = grow & (NN - 1);         // position
        outp[((size_t)bb * DD + gcol) * NN + nn] = (bf16_t)vl;
      }
    }
  }
}

// ---------------------------------------------------------------------------
// Kernel 4: fused scores + bias + softmax + A@V for one 16-row strip.
// Block = 256 threads (8 waves).  LDS: S[16][2048] fp32 (128 KB; CDNA5 WGP
// has 320 KB LDS so 2 blocks co-reside) + small reduction scratch.
// Phase 1: wave w computes S cols [w*256, +256) via 16 WMMA accumulators;
//          B fragments rotate through a 3-deep pipeline and their source
//          lines are prefetched (global_prefetch_b8) 2 k-steps ahead.
// Phase 2: row max + exp (UNNORMALIZED) + rowsum; 1/rowsum kept in LDS.
// Phase 3: wave w computes out cols [w*128, +128): A (LDS fp32 -> bf16 on the
//          fly) x v^T via WMMA, K = 2048; normalization folded into epilogue.
// ---------------------------------------------------------------------------
__global__ void __launch_bounds__(256)
attn_kernel(const bf16_t* __restrict__ qb, const bf16_t* __restrict__ kb,
            const bf16_t* __restrict__ vT, const float* __restrict__ P,
            const float* __restrict__ cbias, float* __restrict__ out) {
  extern __shared__ char smem_raw[];
  float* S    = (float*)smem_raw;       // 16 * 2048
  float* red  = S + 16 * NN;            // 256
  float* rowv = red + 256;              // 16  (row max, then 1/rowsum)

  int tid  = threadIdx.x;
  int lane = tid & 31;
  int wave = tid >> 5;
  int rt = blockIdx.x;                  // row tile 0..127
  int b  = blockIdx.y;                  // batch
  int grow0 = rt * 16;                  // row base within N
  const bf16_t* qs = qb + ((size_t)b * NN + grow0) * DD;
  const bf16_t* ks = kb + (size_t)b * NN * DD;
  const bf16_t* vs = vT + (size_t)b * DD * NN;

  // ---- Phase 1: S = q k^T * scale + P + relbias -> LDS ----
  {
    v8f acc[16] = {};
    int cbase = wave * 256;
    v16bf a   = load_a_bf16(qs, DD, 0, lane);
    v16bf an  = load_a_bf16(qs, DD, 32, lane);
    v16bf bb0 = load_b_bf16(ks + (size_t)(cbase + 0)  * DD, DD, 0, lane);
    v16bf bb1 = load_b_bf16(ks + (size_t)(cbase + 16) * DD, DD, 0, lane);
    for (int k0 = 0; k0 < DD; k0 += 32) {
      int kn  = (k0 + 32 < DD) ? (k0 + 32) : k0;
      int kn2 = (k0 + 64 < DD) ? (k0 + 64) : kn;
#pragma unroll
      for (int j = 0; j < 16; ++j) {
        // prefetch this fragment's source lines 2 k-steps ahead
        __builtin_prefetch(baddr(ks + (size_t)(cbase + j * 16) * DD, DD, kn2, lane), 0, 1);
        v16bf bn;
        if (j + 2 < 16)
          bn = load_b_bf16(ks + (size_t)(cbase + (j + 2) * 16) * DD, DD, k0, lane);
        else
          bn = load_b_bf16(ks + (size_t)(cbase + (j - 14) * 16) * DD, DD, kn, lane);
        acc[j] = wmma_bf16(a, bb0, acc[j]);
        bb0 = bb1; bb1 = bn;
      }
      a  = an;
      an = load_a_bf16(qs, DD, kn2, lane);
    }

    const float scale = 0.03125f;       // 1/sqrt(1024)
    int n  = lane & 15;
    int mb = (lane >> 4) * 8;
#pragma unroll
    for (int j = 0; j < 16; ++j) {
      int col = cbase + j * 16 + n;
#pragma unroll
      for (int r = 0; r < 8; ++r) {
        int mr = mb + r;
        int gi = grow0 + mr;
        float e = acc[j][r] * scale + P[(size_t)gi * NN + col] +
                  cbias[(gi - col) & (TWO_N - 1)];
        S[mr * NN + col] = e;
      }
    }
  }
  __syncthreads();

  // ---- Phase 2: row max, unnormalized exp in LDS, 1/rowsum in rowv ----
  {
    int r   = tid & 15;                 // row in strip
    int seg = tid >> 4;                 // 0..15, 128 cols each
    float* row = S + r * NN;
    int c0 = seg * 128;

    float mx = -3.402823466e38f;
    for (int i = 0; i < 128; ++i) mx = fmaxf(mx, row[c0 + i]);
    red[r * 16 + seg] = mx;
    __syncthreads();
    if (tid < 16) {
      float m = red[tid * 16];
      for (int s2 = 1; s2 < 16; ++s2) m = fmaxf(m, red[tid * 16 + s2]);
      rowv[tid] = m;
    }
    __syncthreads();
    float rm = rowv[r];

    float sum = 0.f;
    for (int i = 0; i < 128; ++i) {
      float p = expf(row[c0 + i] - rm);
      row[c0 + i] = p;                  // unnormalized probability
      sum += p;
    }
    red[r * 16 + seg] = sum;
    __syncthreads();
    if (tid < 16) {
      float s = 0.f;
      for (int s2 = 0; s2 < 16; ++s2) s += red[tid * 16 + s2];
      rowv[tid] = 1.0f / s;             // normalization folded into phase 3
    }
  }
  __syncthreads();

  // ---- Phase 3: out = diag(1/rowsum) * (expE @ v) ----
  {
    auto afrag = [&](int k0) -> v16bf {
      int m   = lane & 15;
      int sub = (lane >> 4) * 16;
      const float* p = S + m * NN + k0;
      v4f c0 = *(const v4f*)(p + sub);
      v4f c1 = *(const v4f*)(p + sub + 4);
      v4f c2 = *(const v4f*)(p + sub + 8);
      v4f c3 = *(const v4f*)(p + sub + 12);
      v16bf a;
#pragma unroll
      for (int i = 0; i < 4; ++i) {
        a[i]      = (bf16_t)c0[i];
        a[4 + i]  = (bf16_t)c1[i];
        a[8 + i]  = (bf16_t)c2[i];
        a[12 + i] = (bf16_t)c3[i];
      }
      return a;
    };

    v8f acc[8] = {};
    int ebase = wave * 128;
    v16bf a   = afrag(0);
    v16bf bb0 = load_b_bf16(vs + (size_t)(ebase + 0)  * NN, NN, 0, lane);
    v16bf bb1 = load_b_bf16(vs + (size_t)(ebase + 16) * NN, NN, 0, lane);
    for (int k0 = 0; k0 < NN; k0 += 32) {
      int kn  = (k0 + 32 < NN) ? (k0 + 32) : k0;
      int kn2 = (k0 + 64 < NN) ? (k0 + 64) : kn;
      v16bf an = afrag(kn);             // LDS latency is short; depth-1 ok
#pragma unroll
      for (int j = 0; j < 8; ++j) {
        __builtin_prefetch(baddr(vs + (size_t)(ebase + j * 16) * NN, NN, kn2, lane), 0, 1);
        v16bf bn;
        if (j + 2 < 8)
          bn = load_b_bf16(vs + (size_t)(ebase + (j + 2) * 16) * NN, NN, k0, lane);
        else
          bn = load_b_bf16(vs + (size_t)(ebase + (j - 6) * 16) * NN, NN, kn, lane);
        acc[j] = wmma_bf16(a, bb0, acc[j]);
        bb0 = bb1; bb1 = bn;
      }
      a = an;
    }

    int n  = lane & 15;
    int mb = (lane >> 4) * 8;
    float rinv[8];
#pragma unroll
    for (int r = 0; r < 8; ++r) rinv[r] = rowv[mb + r];
    float* ob = out + ((size_t)b * NN + grow0) * DD;
#pragma unroll
    for (int j = 0; j < 8; ++j) {
      int e = ebase + j * 16 + n;
#pragma unroll
      for (int r = 0; r < 8; ++r) {
        ob[(size_t)(mb + r) * DD + e] = acc[j][r] * rinv[r];
      }
    }
  }
}

// ---------------------------------------------------------------------------
// Launch
// ---------------------------------------------------------------------------
extern "C" void kernel_launch(void* const* d_in, const int* in_sizes, int n_in,
                              void* d_out, int out_size, void* d_ws, size_t ws_size,
                              hipStream_t stream) {
  const float* x    = (const float*)d_in[0];
  const float* W_Q  = (const float*)d_in[1];
  const float* W_K  = (const float*)d_in[2];
  const float* W_V  = (const float*)d_in[3];
  const float* P    = (const float*)d_in[4];
  const float* R    = (const float*)d_in[5];
  const float* tauQ = (const float*)d_in[6];
  const float* tauK = (const float*)d_in[7];
  const float* tauV = (const float*)d_in[8];
  float* out = (float*)d_out;

  char* ws = (char*)d_ws;
  size_t off = 0;
  float* cbias = (float*)(ws + off);  off += (size_t)TWO_N * sizeof(float);
  off = (off + 255) & ~(size_t)255;
  bf16_t* WtQ = (bf16_t*)(ws + off);  off += (size_t)DD * DD * sizeof(bf16_t);
  bf16_t* WtK = (bf16_t*)(ws + off);  off += (size_t)DD * DD * sizeof(bf16_t);
  bf16_t* WtV = (bf16_t*)(ws + off);  off += (size_t)DD * DD * sizeof(bf16_t);
  bf16_t* xb  = (bf16_t*)(ws + off);  off += (size_t)NB * NN * DD * sizeof(bf16_t);
  bf16_t* qb  = (bf16_t*)(ws + off);  off += (size_t)NB * NN * DD * sizeof(bf16_t);
  bf16_t* kb  = (bf16_t*)(ws + off);  off += (size_t)NB * NN * DD * sizeof(bf16_t);
  bf16_t* vT  = (bf16_t*)(ws + off);  off += (size_t)NB * NN * DD * sizeof(bf16_t);
  (void)ws_size; (void)in_sizes; (void)n_in; (void)out_size;

  // 1) circulant bias column
  dft_kernel<<<TWO_N / 256, 256, 0, stream>>>(R, cbias);

  // 2) weight transpose + bf16 convert; x -> bf16
  wtrans_kernel<<<(DD * DD) / 256, 256, 0, stream>>>(W_Q, WtQ);
  wtrans_kernel<<<(DD * DD) / 256, 256, 0, stream>>>(W_K, WtK);
  wtrans_kernel<<<(DD * DD) / 256, 256, 0, stream>>>(W_V, WtV);
  xconv_kernel<<<((size_t)NB * NN * DD / 4) / 256, 256, 0, stream>>>(x, xb);

  // 3) projections: q, k row-major bf16; v transposed per batch
  dim3 g_qkv(DD / 64, (NB * NN) / 64);
  qkv_kernel<<<g_qkv, 128, 0, stream>>>(xb, WtQ, tauQ, qb, 0);
  qkv_kernel<<<g_qkv, 128, 0, stream>>>(xb, WtK, tauK, kb, 0);
  qkv_kernel<<<g_qkv, 128, 0, stream>>>(xb, WtV, tauV, vT, 1);

  // 4) fused scores + softmax + A@V
  size_t smem = (size_t)(16 * NN + 256 + 16) * sizeof(float);
  dim3 g_attn(NN / 16, NB);
  attn_kernel<<<g_attn, 256, smem, stream>>>(qb, kb, vT, P, cbias, out);
}
